// AttentionRNN_9277129359748
// MI455X (gfx1250) — matrix operand order repaired and checked by reading, not compiled, not run on hardware
//
#include <hip/hip_runtime.h>
#include <hip/hip_bf16.h>

// ---------------------------------------------------------------------------
// AttentionRNN for MI455X (gfx1250): all GEMMs via v_wmma_f32_16x16x32_f16,
// wave32, one wave per 16x64 f32 output tile (A-tile reused across 4 WMMAs),
// f32 accumulate, f16 operands. K is a compile-time template parameter so the
// K-loop fully unrolls and loads pipeline ahead of the WMMAs.
// ---------------------------------------------------------------------------

typedef _Float16 half_t;
typedef __attribute__((ext_vector_type(16))) _Float16 v16h;
typedef __attribute__((ext_vector_type(8)))  _Float16 v8h;
typedef __attribute__((ext_vector_type(8)))  float    v8f;

#define BB   128   // batch
#define TT   64    // seq len
#define HH   256   // hidden
#define VV   128   // vocab
#define G3H  768   // 3*H

// ------------------------------ WMMA helpers --------------------------------
// A tile: 16x32 f16 of row-major A (lda in elements).
// Lane m=lane&15 holds row m0+m; lanes>=16 take K = k0+16..k0+31.
__device__ __forceinline__ v16h load_a_tile(const half_t* A, int lda,
                                            int m0, int k0, int lane) {
  const int mr   = lane & 15;
  const int roff = (lane >> 4) << 4;             // 0 or 16
  const half_t* p = A + (size_t)(m0 + mr) * lda + (k0 + roff);
  v8h lo = *(const v8h*)(p);                     // K = k0+roff .. +7
  v8h hi = *(const v8h*)(p + 8);                 // K = k0+roff+8 .. +15
  v16h a;
#pragma unroll
  for (int i = 0; i < 8; ++i) { a[i] = lo[i]; a[8 + i] = hi[i]; }
  return a;
}

// B tile (K x 16) taken as W^T where W is row-major (N x K):
// lane n=lane&15 loads 16 contiguous halves of row n0+n at k0+(lane>=16?16:0).
__device__ __forceinline__ v16h load_bt_tile(const half_t* W, int ldw,
                                             int n0, int k0, int lane) {
  const int nc   = lane & 15;
  const int koff = (lane >> 4) << 4;             // 0 or 16
  return *(const v16h*)(W + (size_t)(n0 + nc) * ldw + (k0 + koff));
}

__device__ __forceinline__ v8f wmma_f16(v16h a, v16h b, v8f c) {
  return __builtin_amdgcn_wmma_f32_16x16x32_f16(
      /*neg_a=*/false, a, /*neg_b=*/false, b,
      /*c_mod=*/(short)0, c, /*reuse_a=*/false, /*reuse_b=*/false);
}

__device__ __forceinline__ float sigmoidf_(float x) {
  return 1.0f / (1.0f + __expf(-x));
}

// ------------------------------ utility kernels -----------------------------
__global__ void cast_f32_to_f16_kernel(const float* __restrict__ src,
                                       half_t* __restrict__ dst, int n) {
  int i = blockIdx.x * blockDim.x + threadIdx.x;
  if (i < n) dst[i] = (half_t)src[i];
}

__global__ void embed_kernel(const int* __restrict__ x,
                             const float* __restrict__ embed,
                             half_t* __restrict__ emb16) {
  // grid = B*T blocks, block = H threads
  int row = blockIdx.x;               // b*T + t
  int h   = threadIdx.x;              // 0..255
  int idx = x[row];
  emb16[(size_t)row * HH + h] = (half_t)embed[(size_t)idx * HH + h];
}

__global__ void init_h_kernel(float* __restrict__ hf, half_t* __restrict__ hh) {
  int i = blockIdx.x * blockDim.x + threadIdx.x;
  if (i < BB * HH) { hf[i] = 0.0f; hh[i] = (half_t)0.0f; }
}

__global__ void copy_f32_kernel(const float* __restrict__ src,
                                float* __restrict__ dst, int n) {
  int i = blockIdx.x * blockDim.x + threadIdx.x;
  if (i < n) dst[i] = src[i];
}

// ------------------------------ generic WMMA GEMM ---------------------------
// C(MxN,f32,ldc) = A(MxK,f16,lda) * W(NxK,f16,ldw)^T + bias(N, optional)
// One wave (32 threads) per 16x64 tile: A loaded once per K-step, 4 B tiles,
// 4 WMMAs.  N multiple of 64; K compile-time (full unroll, no branches).
template <int K>
__global__ void wmma_gemm_bias_kernel(const half_t* __restrict__ A, int lda,
                                      const half_t* __restrict__ W, int ldw,
                                      const float* __restrict__ bias,
                                      float* __restrict__ C, int ldc, int N) {
  const int lane = threadIdx.x;
  const int ntiles = N >> 6;                      // 64-wide column tiles
  const int m0 = (blockIdx.x / ntiles) << 4;
  const int n0 = (blockIdx.x % ntiles) << 6;

  v8f acc0 = {}, acc1 = {}, acc2 = {}, acc3 = {};
#pragma unroll
  for (int k0 = 0; k0 < K; k0 += 32) {
    if (k0 + 32 < K) {  // compile-time condition: no branch after unroll
      __builtin_prefetch(A + (size_t)(m0 + (lane & 15)) * lda + k0 + 32, 0, 3);
      __builtin_prefetch(W + (size_t)(n0 + (lane & 15)) * ldw + k0 + 32, 0, 3);
    }
    v16h a  = load_a_tile(A, lda, m0, k0, lane);
    v16h b0 = load_bt_tile(W, ldw, n0,      k0, lane);
    v16h b1 = load_bt_tile(W, ldw, n0 + 16, k0, lane);
    v16h b2 = load_bt_tile(W, ldw, n0 + 32, k0, lane);
    v16h b3 = load_bt_tile(W, ldw, n0 + 48, k0, lane);
    acc0 = wmma_f16(a, b0, acc0);
    acc1 = wmma_f16(a, b1, acc1);
    acc2 = wmma_f16(a, b2, acc2);
    acc3 = wmma_f16(a, b3, acc3);
  }

  const int nc    = lane & 15;
  const int rbase = m0 + ((lane >> 4) << 3);
  const v8f accs[4] = {acc0, acc1, acc2, acc3};
#pragma unroll
  for (int s = 0; s < 4; ++s) {
    const int col  = n0 + s * 16 + nc;
    const float bv = bias ? bias[col] : 0.0f;
#pragma unroll
    for (int i = 0; i < 8; ++i)
      C[(size_t)(rbase + i) * ldc + col] = accs[s][i] + bv;
  }
}

// ------------------------------ GRU recurrent step --------------------------
// One wave computes a 16x16 (batch x hidden) tile of h_new.
// hp = h @ w_hh^T + b_hh  (three gate slabs of the 768-wide projection),
// then gates:  r=sig(xr+hr)  z=sig(xz+hz)  n=tanh(xn+r*hn)
//              h_new = (1-z)*n + z*h
__global__ void gru_step_kernel(const half_t* __restrict__ hin16,
                                const float*  __restrict__ hinf,
                                const half_t* __restrict__ whh16,  // (768x256)
                                const float*  __restrict__ bhh,    // (768)
                                const float*  __restrict__ xpt,    // xp + t*768
                                int xp_bstride,                    // T*768
                                half_t* __restrict__ hout16,
                                float*  __restrict__ houtf,
                                half_t* __restrict__ yout,         // base + t*ld
                                int y_bstride) {                   // T*ld
  const int lane = threadIdx.x;
  const int ntiles = HH >> 4;                     // 16
  const int m0 = (blockIdx.x / ntiles) << 4;      // batch tile
  const int n0 = (blockIdx.x % ntiles) << 4;      // hidden tile

  v8f cr = {}, cz = {}, cn = {};
#pragma unroll
  for (int k0 = 0; k0 < HH; k0 += 32) {
    v16h a  = load_a_tile(hin16, HH, m0, k0, lane);
    v16h br = load_bt_tile(whh16,                 HH, n0, k0, lane);
    v16h bz = load_bt_tile(whh16 + 256 * HH,      HH, n0, k0, lane);
    v16h bn = load_bt_tile(whh16 + 512 * HH,      HH, n0, k0, lane);
    cr = wmma_f16(a, br, cr);
    cz = wmma_f16(a, bz, cz);
    cn = wmma_f16(a, bn, cn);
  }

  const int col   = n0 + (lane & 15);
  const int rbase = m0 + ((lane >> 4) << 3);
  const float bhr = bhh[col];
  const float bhz = bhh[256 + col];
  const float bhn = bhh[512 + col];
#pragma unroll
  for (int i = 0; i < 8; ++i) {
    const int b = rbase + i;
    const float* xr = xpt + (size_t)b * xp_bstride;
    const float xrv = xr[col];
    const float xzv = xr[256 + col];
    const float xnv = xr[512 + col];
    const float hr = cr[i] + bhr;
    const float hz = cz[i] + bhz;
    const float hn = cn[i] + bhn;
    const float r = sigmoidf_(xrv + hr);
    const float z = sigmoidf_(xzv + hz);
    const float n = tanhf(xnv + r * hn);
    const float hp = hinf[(size_t)b * HH + col];
    const float hnew = (1.0f - z) * n + z * hp;
    houtf[(size_t)b * HH + col]  = hnew;
    hout16[(size_t)b * HH + col] = (half_t)hnew;
    yout[(size_t)b * y_bstride + col] = (half_t)hnew;
  }
}

// --------------------- attention energy + softmax (per b,t) -----------------
// energy[b,t,k] = sum_h v_a[h] * tanh(Wa_e[b,k,h] + Ua_h[b,t,h]); softmax over k.
// Block = 256 threads (8 waves); wave w handles k = 8w..8w+7, lanes stride h.
__global__ void attn_softmax_kernel(const float* __restrict__ Wa_e,
                                    const float* __restrict__ Ua_h,
                                    const float* __restrict__ v_a,
                                    half_t* __restrict__ alpha16) {
  const int b = blockIdx.x / TT;
  const int t = blockIdx.x % TT;
  const int tid  = threadIdx.x;
  const int wave = tid >> 5;
  const int lane = tid & 31;

  float uq[8], va[8];
#pragma unroll
  for (int j = 0; j < 8; ++j) {
    const int h = lane + 32 * j;
    uq[j] = Ua_h[((size_t)b * TT + t) * HH + h];
    va[j] = v_a[h];
  }

  __shared__ float erow[TT];
  const float* wbase = Wa_e + (size_t)b * TT * HH;
#pragma unroll
  for (int kk = 0; kk < 8; ++kk) {
    const int k = wave * 8 + kk;
    const float* wrow = wbase + (size_t)k * HH;
    float s = 0.0f;
#pragma unroll
    for (int j = 0; j < 8; ++j)
      s += va[j] * tanhf(wrow[lane + 32 * j] + uq[j]);
#pragma unroll
    for (int off = 16; off > 0; off >>= 1)
      s += __shfl_xor(s, off, 32);
    if (lane == 0) erow[k] = s;
  }
  __syncthreads();

  if (wave == 0) {
    float e0 = erow[lane], e1 = erow[lane + 32];
    float m = fmaxf(e0, e1);
#pragma unroll
    for (int off = 16; off > 0; off >>= 1)
      m = fmaxf(m, __shfl_xor(m, off, 32));
    float p0 = __expf(e0 - m), p1 = __expf(e1 - m);
    float s = p0 + p1;
#pragma unroll
    for (int off = 16; off > 0; off >>= 1)
      s += __shfl_xor(s, off, 32);
    const float inv = 1.0f / s;
    half_t* arow = alpha16 + ((size_t)b * TT + t) * TT;
    arow[lane]      = (half_t)(p0 * inv);
    arow[lane + 32] = (half_t)(p1 * inv);
  }
}

// --------------------- context GEMM: ctx = alpha @ enc ----------------------
// Per batch: (64x64 f16) @ (64x256 f16, enc is cols 0..255 of combined/512).
// Writes f16 ctx into combined cols 256..511. One wave per 16x64 tile.
__global__ void ctx_gemm_kernel(const half_t* __restrict__ alpha16,
                                half_t* __restrict__ combined) {
  const int lane  = threadIdx.x;
  const int tiles = (TT / 16) * (HH / 64);        // 4*4 = 16
  const int b  = blockIdx.x / tiles;
  const int r  = blockIdx.x % tiles;
  const int m0 = (r / (HH / 64)) << 4;            // t tile
  const int n0 = (r % (HH / 64)) << 6;            // h tile (64-wide)

  const half_t* A = alpha16 + (size_t)b * TT * TT;          // lda = 64
  const half_t* E = combined + (size_t)b * TT * (2 * HH);   // enc rows, ld 512

  v8f acc0 = {}, acc1 = {}, acc2 = {}, acc3 = {};
  const int nc   = lane & 15;
  const int koff = (lane >> 4) << 4;
#pragma unroll
  for (int k0 = 0; k0 < TT; k0 += 32) {
    v16h a = load_a_tile(A, TT, m0, k0, lane);
    // B tiles: B[k][n] = enc[b, k, n]  (row-major KxN view, ld = 512)
    v16h b0, b1, b2, b3;
#pragma unroll
    for (int kk = 0; kk < 16; ++kk) {
      const half_t* er = E + (size_t)(k0 + koff + kk) * (2 * HH) + n0 + nc;
      b0[kk] = er[0];
      b1[kk] = er[16];
      b2[kk] = er[32];
      b3[kk] = er[48];
    }
    acc0 = wmma_f16(a, b0, acc0);
    acc1 = wmma_f16(a, b1, acc1);
    acc2 = wmma_f16(a, b2, acc2);
    acc3 = wmma_f16(a, b3, acc3);
  }

  const int rbase = m0 + ((lane >> 4) << 3);
  const v8f accs[4] = {acc0, acc1, acc2, acc3};
#pragma unroll
  for (int s = 0; s < 4; ++s) {
    const int col = n0 + s * 16 + nc;
#pragma unroll
    for (int i = 0; i < 8; ++i)
      combined[((size_t)b * TT + rbase + i) * (2 * HH) + HH + col] =
          (half_t)accs[s][i];
  }
}

// ------------------------------ host launcher -------------------------------
extern "C" void kernel_launch(void* const* d_in, const int* in_sizes, int n_in,
                              void* d_out, int out_size, void* d_ws, size_t ws_size,
                              hipStream_t stream) {
  (void)in_sizes; (void)n_in; (void)out_size; (void)ws_size;

  const int*   x      = (const int*)d_in[0];
  const float* embed  = (const float*)d_in[1];
  const float* w_ih0  = (const float*)d_in[2];
  const float* w_hh0  = (const float*)d_in[3];
  const float* b_ih0  = (const float*)d_in[4];
  const float* b_hh0  = (const float*)d_in[5];
  const float* w_ih1  = (const float*)d_in[6];
  const float* w_hh1  = (const float*)d_in[7];
  const float* b_ih1  = (const float*)d_in[8];
  const float* b_hh1  = (const float*)d_in[9];
  const float* W_a    = (const float*)d_in[10];
  const float* U_a    = (const float*)d_in[11];
  const float* v_a    = (const float*)d_in[12];
  const float* fc_W   = (const float*)d_in[13];
  const float* fc_b   = (const float*)d_in[14];
  float* out = (float*)d_out;

  // ---- workspace layout (all offsets 256B aligned) ----
  char* ws = (char*)d_ws;
  size_t off = 0;
  auto alloc = [&](size_t bytes) { char* p = ws + off; off += (bytes + 255) & ~(size_t)255; return p; };

  half_t* w_ih0_16 = (half_t*)alloc(G3H * HH * 2);
  half_t* w_hh0_16 = (half_t*)alloc(G3H * HH * 2);
  half_t* w_ih1_16 = (half_t*)alloc(G3H * HH * 2);
  half_t* w_hh1_16 = (half_t*)alloc(G3H * HH * 2);
  half_t* W_a16    = (half_t*)alloc(HH * HH * 2);
  half_t* U_a16    = (half_t*)alloc(HH * HH * 2);
  half_t* fc_W16   = (half_t*)alloc(VV * 2 * HH * 2);
  half_t* emb16    = (half_t*)alloc((size_t)BB * TT * HH * 2);
  half_t* y0_16    = (half_t*)alloc((size_t)BB * TT * HH * 2);     // h1_seq
  half_t* combined = (half_t*)alloc((size_t)BB * TT * 2 * HH * 2); // [enc|ctx]
  half_t* alpha16  = (half_t*)alloc((size_t)BB * TT * TT * 2);
  float*  hfA      = (float*)alloc(BB * HH * 4);
  float*  hfB      = (float*)alloc(BB * HH * 4);
  half_t* hhA      = (half_t*)alloc(BB * HH * 2);
  half_t* hhB      = (half_t*)alloc(BB * HH * 2);
  float*  xp       = (float*)alloc((size_t)BB * TT * G3H * 4);     // reused
  float*  Wa_e     = xp;                                           // reuse xp
  float*  Ua_h     = xp + (size_t)BB * TT * HH;                    // after layer1

  const int M = BB * TT;  // 8192

  // ---- 1) cast all weights to f16 ----
  struct { const float* s; half_t* d; int n; } casts[] = {
    {w_ih0, w_ih0_16, G3H * HH}, {w_hh0, w_hh0_16, G3H * HH},
    {w_ih1, w_ih1_16, G3H * HH}, {w_hh1, w_hh1_16, G3H * HH},
    {W_a,   W_a16,    HH * HH},  {U_a,   U_a16,    HH * HH},
    {fc_W,  fc_W16,   VV * 2 * HH},
  };
  for (auto& c : casts)
    cast_f32_to_f16_kernel<<<(c.n + 255) / 256, 256, 0, stream>>>(c.s, c.d, c.n);

  // ---- 2) embedding gather (f16) ----
  embed_kernel<<<M, HH, 0, stream>>>(x, embed, emb16);

  // ---- 3) layer-0 input projection: xp = emb @ w_ih0^T + b_ih0 ----
  wmma_gemm_bias_kernel<HH><<<(M / 16) * (G3H / 64), 32, 0, stream>>>(
      emb16, HH, w_ih0_16, HH, b_ih0, xp, G3H, G3H);

  // ---- 4) layer-0 recurrence ----
  init_h_kernel<<<(BB * HH + 255) / 256, 256, 0, stream>>>(hfA, hhA);
  for (int t = 0; t < TT; ++t) {
    const bool odd = t & 1;
    gru_step_kernel<<<(BB / 16) * (HH / 16), 32, 0, stream>>>(
        odd ? hhB : hhA, odd ? hfB : hfA, w_hh0_16, b_hh0,
        xp + (size_t)t * G3H, TT * G3H,
        odd ? hhA : hhB, odd ? hfA : hfB,
        y0_16 + (size_t)t * HH, TT * HH);
  }

  // ---- 5) layer-1 input projection: xp = y0 @ w_ih1^T + b_ih1 ----
  wmma_gemm_bias_kernel<HH><<<(M / 16) * (G3H / 64), 32, 0, stream>>>(
      y0_16, HH, w_ih1_16, HH, b_ih1, xp, G3H, G3H);

  // ---- 6) layer-1 recurrence; enc written into combined[:, 0:256] ----
  init_h_kernel<<<(BB * HH + 255) / 256, 256, 0, stream>>>(hfA, hhA);
  for (int t = 0; t < TT; ++t) {
    const bool odd = t & 1;
    gru_step_kernel<<<(BB / 16) * (HH / 16), 32, 0, stream>>>(
        odd ? hhB : hhA, odd ? hfB : hfA, w_hh1_16, b_hh1,
        xp + (size_t)t * G3H, TT * G3H,
        odd ? hhA : hhB, odd ? hfA : hfB,
        combined + (size_t)t * 2 * HH, TT * 2 * HH);
  }
  // final hidden (t=63 wrote into A-buffers)
  copy_f32_kernel<<<(BB * HH + 255) / 256, 256, 0, stream>>>(
      hfA, out + (size_t)M * VV, BB * HH);

  // ---- 7) attention projections (xp buffer now free, reuse as Wa_e/Ua_h) ----
  wmma_gemm_bias_kernel<HH><<<(M / 16) * (HH / 64), 32, 0, stream>>>(
      combined, 2 * HH, W_a16, HH, nullptr, Wa_e, HH, HH);
  wmma_gemm_bias_kernel<HH><<<(M / 16) * (HH / 64), 32, 0, stream>>>(
      combined, 2 * HH, U_a16, HH, nullptr, Ua_h, HH, HH);

  // ---- 8) energy + softmax -> alpha16 ----
  attn_softmax_kernel<<<BB * TT, 256, 0, stream>>>(Wa_e, Ua_h, v_a, alpha16);

  // ---- 9) ctx = alpha @ enc -> combined[:, 256:512] ----
  ctx_gemm_kernel<<<BB * (TT / 16) * (HH / 64), 32, 0, stream>>>(
      alpha16, combined);

  // ---- 10) logits = combined @ fc_W^T + fc_b -> d_out ----
  wmma_gemm_bias_kernel<2 * HH><<<(M / 16) * (VV / 64), 32, 0, stream>>>(
      combined, 2 * HH, fc_W16, 2 * HH, fc_b, out, VV, VV);
}
